// DecoderAttn_33079838114120
// MI455X (gfx1250) — compile-verified, matrix operand order checked by reading, not run on hardware
//
#include <hip/hip_runtime.h>
#include <hip/hip_bf16.h>

#define DEV __device__ __forceinline__

typedef _Float16 v16h __attribute__((ext_vector_type(16)));
typedef float    v8f  __attribute__((ext_vector_type(8)));
typedef unsigned uv4  __attribute__((ext_vector_type(4)));
typedef int      iv8  __attribute__((ext_vector_type(8)));
typedef int      iv4  __attribute__((ext_vector_type(4)));

#if __has_builtin(__builtin_amdgcn_tensor_load_to_lds) && \
    __has_builtin(__builtin_amdgcn_s_wait_tensorcnt)
#define USE_TDM 1
#else
#define USE_TDM 0
#endif

namespace {

constexpr int Bb  = 4;
constexpr int Nn  = 2048;
constexpr int Dd  = 384;
constexpr int Hh  = 8;
constexpr int Gg  = 512;
constexpr int Kn  = 8;            // knn neighbors
constexpr int HID = 4 * Dd;       // 1536
constexpr int FIN = 3 + Gg;       // 515
constexpr int FIN_KP = 544;       // FIN padded to mult of 32 (K dim)
constexpr int FIN_NP = 576;       // FIN padded to mult of 64 (N dim)
constexpr int M   = Bb * Nn;      // 8192 rows
constexpr int HD  = Dd / Hh;      // 48 head dim
constexpr int CH  = 4;            // K-steps per TDM chunk (16KB per buffer)

DEV v8f wmma16(v16h a, v16h b, v8f c) {
  // D = A(16x32 f16) * B(32x16 f16) + C(16x16 f32)
  return __builtin_amdgcn_wmma_f32_16x16x32_f16(false, a, false, b, (short)0, c,
                                                false, false);
}

DEV int frag_k(int lane, int h) {
  // 16-bit A/B fragment: halves 0..7 -> K base 0 (lanes 0-15) / 8 (lanes 16-31)
  //                      halves 8..15 -> K base 16 / 24
  int hh = h & 7;
  if (h < 8) return ((lane < 16) ? 0 : 8) + hh;
  return ((lane < 16) ? 16 : 24) + hh;
}

DEV float gelu_exact(float x) {
  return 0.5f * x * (1.f + erff(x * 0.70710678118654752f));
}

// ---- fragment loaders -------------------------------------------------------

DEV v16h load_a_frag(const _Float16* __restrict__ A, int lda, int row, int k0,
                     int lane) {
  const _Float16* p = A + (size_t)row * lda + k0 + ((lane < 16) ? 0 : 8);
  union { v16h v; uint4 u[2]; } t;
  t.u[0] = *(const uint4*)p;
  t.u[1] = *(const uint4*)(p + 16);
  return t.v;
}

DEV v16h load_frag32B(const _Float16* p) {  // 16 contiguous halves
  union { v16h v; uint4 u[2]; } t;
  t.u[0] = ((const uint4*)p)[0];
  t.u[1] = ((const uint4*)p)[1];
  return t.v;
}

DEV v16h load_wp_frag(const _Float16* __restrict__ Wp, int KB, int nblk,
                      int kblk, int lane) {
  return load_frag32B(Wp + ((size_t)(nblk * KB + kblk) * 512 + lane * 16));
}

#if USE_TDM
// TDM: DMA `rem` K-steps of packed-B (4 nblk rows x rem*512 halves, row stride
// KB*512 halves) from global into LDS. Rows packed contiguously in LDS.
// Issued once per wave (EXEC ignored; scalar descriptor in SGPRs).
DEV void tdm_load_b(const _Float16* gsrc, unsigned lds_byte_off, int KB,
                    int rem) {
  uint64_t ga = (uint64_t)(uintptr_t)gsrc;
  unsigned tile0 = (unsigned)rem * 512u;       // halves per row
  uv4 g0;
  g0[0] = 1u;                                  // count=1, user descriptor
  g0[1] = lds_byte_off;                        // LDS destination (bytes)
  g0[2] = (unsigned)(ga & 0xffffffffu);        // global_addr[31:0]
  g0[3] = (unsigned)((ga >> 32) & 0x01ffffffu) // global_addr[56:32]
          | (2u << 30);                        // type = 2 (image/tensor)
  unsigned stride0 = (unsigned)KB * 512u;      // elements (halves)
  iv8 g1;
  g1[0] = (int)(1u << 16);                     // data_size=1 (2 bytes)
  g1[1] = (int)(tile0 << 16);                  // tensor_dim0[15:0]
  g1[2] = (int)((tile0 >> 16) | (4u << 16));   // dim0 hi | tensor_dim1=4
  g1[3] = (int)(tile0 << 16);                  // dim1 hi=0 | tile_dim0
  g1[4] = 4;                                   // tile_dim1=4, tile_dim2=0
  g1[5] = (int)stride0;                        // tensor_dim0_stride[31:0]
  g1[6] = 0;                                   // stride0 hi | stride1 lo
  g1[7] = 0;
  iv4 z4 = {0, 0, 0, 0};
  iv8 z8 = {0, 0, 0, 0, 0, 0, 0, 0};
  __builtin_amdgcn_tensor_load_to_lds(g0, g1, z4, z4, z8, 0);
}
#endif

// ---- weight packing: W[K,Nc] fp32 -> f16 B-fragments [Np/16][Kp/32][32][16] --

__global__ void k_pack_w(const float* __restrict__ W, _Float16* __restrict__ Wp,
                         int K, int Nc, int Kp, int Np) {
  int i = blockIdx.x * blockDim.x + threadIdx.x;
  int KB = Kp >> 5;
  int total = (Np >> 4) * KB * 512;
  if (i >= total) return;
  int h = i & 15, lane = (i >> 4) & 31, blk = i >> 9;
  int kblk = blk % KB, nblk = blk / KB;
  int k = kblk * 32 + frag_k(lane, h);
  int n = nblk * 16 + (lane & 15);
  float v = (k < K && n < Nc) ? W[(size_t)k * Nc + n] : 0.f;
  Wp[i] = (_Float16)v;
}

// ---- generic WMMA GEMM: C = act(A @ W + bias) -------------------------------
// block: 256 thr = 8 waves; block tile 128 rows x 64 cols; wave tile 16x64.
// B operands staged through LDS by the Tensor Data Mover, double buffered,
// CH K-steps per DMA chunk (one barrier per CH K-steps).

__global__ __launch_bounds__(256) void k_gemm(
    const _Float16* __restrict__ A, int lda, const _Float16* __restrict__ Wp,
    int KB, const float* __restrict__ bias, float* __restrict__ Cf,
    _Float16* __restrict__ Ch, int ldc, int col_off, int Mrows, int Nreal,
    int act) {
  int lane = threadIdx.x & 31;
  int wave = threadIdx.x >> 5;
  int row_base = blockIdx.y * 128 + wave * 16;
  int nb_base = blockIdx.x * 4;
  int arow = row_base + (lane & 15);
  v8f acc[4] = {};

#if USE_TDM
  __shared__ __align__(16) _Float16 sB[2][4 * CH * 512];  // 2 x 16KB
  const _Float16* wbase = Wp + (size_t)nb_base * KB * 512;
  unsigned lds0 = (unsigned)(uintptr_t)&sB[0][0];
  unsigned lds1 = (unsigned)(uintptr_t)&sB[1][0];
  int nch = (KB + CH - 1) / CH;
  int rem0 = (KB < CH) ? KB : CH;
  if (wave == 0) tdm_load_b(wbase, lds0, KB, rem0);  // chunk 0 -> buf 0
  for (int c = 0; c < nch; ++c) {
    int rem = KB - c * CH;
    if (rem > CH) rem = CH;
    if (wave == 0) __builtin_amdgcn_s_wait_tensorcnt(0);
    __syncthreads();  // chunk c visible to all waves; prior reads done
    if (wave == 0 && c + 1 < nch) {
      int remn = KB - (c + 1) * CH;
      if (remn > CH) remn = CH;
      tdm_load_b(wbase + (size_t)(c + 1) * CH * 512,
                 ((c + 1) & 1) ? lds1 : lds0, KB, remn);
    }
    const _Float16* base = &sB[c & 1][0];
    int cw = rem * 512;  // LDS row stride (halves) for this chunk
    for (int j = 0; j < rem; ++j) {
      int kb = c * CH + j;
      if (kb + 1 < KB)  // prefetch next A cacheline (global_prefetch_b8)
        __builtin_prefetch(A + (size_t)arow * lda + (kb + 1) * 32, 0, 1);
      v16h a = load_a_frag(A, lda, arow, kb * 32, lane);
      v16h bf[4];
#pragma unroll
      for (int t = 0; t < 4; ++t)
        bf[t] = load_frag32B(base + t * cw + j * 512 + lane * 16);
#pragma unroll
      for (int t = 0; t < 4; ++t) acc[t] = wmma16(a, bf[t], acc[t]);
    }
  }
#else
  for (int kb = 0; kb < KB; ++kb) {
    v16h a = load_a_frag(A, lda, arow, kb * 32, lane);
    v16h bf[4];
#pragma unroll
    for (int t = 0; t < 4; ++t)
      bf[t] = load_wp_frag(Wp, KB, nb_base + t, kb, lane);
#pragma unroll
    for (int t = 0; t < 4; ++t) acc[t] = wmma16(a, bf[t], acc[t]);
  }
#endif

  int half_off = (lane < 16) ? 0 : 8;
#pragma unroll
  for (int t = 0; t < 4; ++t) {
    int col = (nb_base + t) * 16 + (lane & 15);
    if (col >= Nreal) continue;
    float bv = bias ? bias[col] : 0.f;
#pragma unroll
    for (int r = 0; r < 8; ++r) {
      int row = row_base + r + half_off;
      if (row >= Mrows) continue;
      float v = acc[t][r] + bv;
      if (act == 1) v = gelu_exact(v);
      size_t o = (size_t)row * ldc + col_off + col;
      if (Cf) Cf[o] = v;
      if (Ch) Ch[o] = (_Float16)v;
    }
  }
}

// ---- flash-style attention (self + cross) -----------------------------------
// grid: (B*H, Nq/128); block 256 (8 waves, 16 queries each). hd = 48.

__global__ __launch_bounds__(256) void k_attn(
    const _Float16* __restrict__ Qm, int q_stride, int q_off,
    const _Float16* __restrict__ Km, int k_stride, int k_off,
    const _Float16* __restrict__ Vm, int v_stride, int v_off,
    _Float16* __restrict__ Out, int o_stride, int Nq, int Nk, float scale) {
  int b = blockIdx.x / Hh;
  int h = blockIdx.x % Hh;
  int lane = threadIdx.x & 31;
  int wave = threadIdx.x >> 5;
  int q0 = blockIdx.y * 128 + wave * 16;

  __shared__ __align__(16) _Float16 sK[32][64];   // keys x hd (padded to 64)
  __shared__ __align__(16) _Float16 sVt[HD][32];  // hd x keys (transposed)
  __shared__ __align__(16) _Float16 sP[8][16][32];

  // Q fragments (hd 0..31 and 32..47 zero-padded to 63)
  v16h aq0, aq1;
  {
    int row = q0 + (lane & 15);
    const _Float16* p =
        Qm + (size_t)(b * Nq + row) * q_stride + q_off + h * HD;
    int c0 = (lane < 16) ? 0 : 8;
    union { v16h v; uint4 u[2]; } t;
    t.u[0] = *(const uint4*)(p + c0);
    t.u[1] = *(const uint4*)(p + c0 + 16);
    aq0 = t.v;
    t.u[0] = *(const uint4*)(p + 32 + c0);  // hd 32..39 / 40..47
    uint4 z; z.x = z.y = z.z = z.w = 0u;
    t.u[1] = z;                              // hd 48..63 pad
    aq1 = t.v;
  }

  float mrow[8], lrow[8];
#pragma unroll
  for (int r = 0; r < 8; ++r) { mrow[r] = -3.0e38f; lrow[r] = 0.f; }
  v8f o0 = {}, o1 = {}, o2 = {};

  for (int kt = 0; kt < Nk; kt += 32) {
    __syncthreads();
    for (int i = threadIdx.x; i < 32 * 64; i += 256) {
      int key = i >> 6, c = i & 63;
      _Float16 val = (_Float16)0.f;
      if (c < HD)
        val = Km[(size_t)(b * Nk + kt + key) * k_stride + k_off + h * HD + c];
      sK[key][c] = val;
    }
    for (int i = threadIdx.x; i < 32 * HD; i += 256) {
      int key = i / HD, c = i % HD;
      sVt[c][key] =
          Vm[(size_t)(b * Nk + kt + key) * v_stride + v_off + h * HD + c];
    }
    __syncthreads();

    // S = Q @ K^T : two 16-key column tiles
    v8f s[2] = {};
#pragma unroll
    for (int t = 0; t < 2; ++t) {
      int key = t * 16 + (lane & 15);
      int c0 = (lane < 16) ? 0 : 8;
      union { v16h v; uint4 u[2]; } lo, hi;
      lo.u[0] = *(const uint4*)&sK[key][c0];
      lo.u[1] = *(const uint4*)&sK[key][c0 + 16];
      hi.u[0] = *(const uint4*)&sK[key][32 + c0];
      hi.u[1] = *(const uint4*)&sK[key][32 + c0 + 16];  // zeros (pad)
      s[t] = wmma16(aq0, lo.v, s[t]);
      s[t] = wmma16(aq1, hi.v, s[t]);
    }

    // online softmax; rows live across 16-lane halves
#pragma unroll
    for (int r = 0; r < 8; ++r) {
      float sv0 = s[0][r] * scale, sv1 = s[1][r] * scale;
      float mx = fmaxf(sv0, sv1);
      for (int mm = 1; mm <= 8; mm <<= 1) mx = fmaxf(mx, __shfl_xor(mx, mm, 32));
      float mn = fmaxf(mrow[r], mx);
      float corr = __expf(mrow[r] - mn);
      float p0 = __expf(sv0 - mn), p1 = __expf(sv1 - mn);
      float ps = p0 + p1;
      for (int mm = 1; mm <= 8; mm <<= 1) ps += __shfl_xor(ps, mm, 32);
      lrow[r] = lrow[r] * corr + ps;
      mrow[r] = mn;
      o0[r] *= corr; o1[r] *= corr; o2[r] *= corr;
      int rowl = r + ((lane < 16) ? 0 : 8);
      sP[wave][rowl][lane & 15] = (_Float16)p0;
      sP[wave][rowl][16 + (lane & 15)] = (_Float16)p1;
    }
    __syncthreads();

    // O += P(16x32) @ V(32x48); V fragments are two ds_load_b128 each
    union { v16h v; uint4 u[2]; } pa;
    {
      int rowl = lane & 15;
      int c0 = (lane < 16) ? 0 : 8;
      pa.u[0] = *(const uint4*)&sP[wave][rowl][c0];
      pa.u[1] = *(const uint4*)&sP[wave][rowl][c0 + 16];
    }
    int kb0 = (lane < 16) ? 0 : 8;
    v16h bv[3];
#pragma unroll
    for (int t = 0; t < 3; ++t) {
      int hdc = t * 16 + (lane & 15);
      union { v16h v; uint4 u[2]; } bb;
      bb.u[0] = *(const uint4*)&sVt[hdc][kb0];
      bb.u[1] = *(const uint4*)&sVt[hdc][16 + kb0];
      bv[t] = bb.v;
    }
    o0 = wmma16(pa.v, bv[0], o0);
    o1 = wmma16(pa.v, bv[1], o1);
    o2 = wmma16(pa.v, bv[2], o2);
  }

  int colb = lane & 15;
  int rowoff = (lane < 16) ? 0 : 8;
#pragma unroll
  for (int r = 0; r < 8; ++r) {
    int row = q0 + r + rowoff;
    float inv = 1.f / lrow[r];
    _Float16* orow = Out + (size_t)(b * Nq + row) * o_stride + h * HD;
    orow[colb] = (_Float16)(o0[r] * inv);
    orow[16 + colb] = (_Float16)(o1[r] * inv);
    orow[32 + colb] = (_Float16)(o2[r] * inv);
  }
}

// ---- knn graph-feature GEMM + leakyReLU + max over K neighbors --------------
// feat = [x_knn - x_q, x_q] (2D=768 wide) @ W(768,384); out f16 into concat buf.

__global__ __launch_bounds__(256) void k_knn(
    const _Float16* __restrict__ Xk, const _Float16* __restrict__ Xq,
    const int* __restrict__ idx, const _Float16* __restrict__ Wp,
    const float* __restrict__ bias, _Float16* __restrict__ Out, int ldc,
    int col_off) {
  int lane = threadIdx.x & 31;
  int wave = threadIdx.x >> 5;
  int row_base = blockIdx.y * 128 + wave * 16;
  int nb_base = blockIdx.x * 4;
  int row = row_base + (lane & 15);
  int n = row % Nn, b = row / Nn;
  const int KB = (2 * Dd) >> 5;  // 24
  const _Float16* xq = Xq + (size_t)row * Dd;

  v8f best[4];
#pragma unroll
  for (int t = 0; t < 4; ++t)
#pragma unroll
    for (int r = 0; r < 8; ++r) best[t][r] = -3.0e38f;

  for (int nb = 0; nb < Kn; ++nb) {
    int gi = idx[((size_t)b * Kn + nb) * Nn + n];
    const _Float16* xk = Xk + (size_t)gi * Dd;
    v8f acc[4] = {};
    for (int kb = 0; kb < KB; ++kb) {
      int c0 = kb * 32 + ((lane < 16) ? 0 : 8);
      union { v16h v; uint4 u[2]; } a;
      // group of 8 contiguous halves of the 768-wide feature
      auto grp = [&](int g) -> uint4 {
        if (g < Dd) {
          union { uint4 q; _Float16 e[8]; } A_, B_;
          A_.q = *(const uint4*)(xk + g);
          B_.q = *(const uint4*)(xq + g);
#pragma unroll
          for (int j = 0; j < 8; ++j) A_.e[j] = (_Float16)(A_.e[j] - B_.e[j]);
          return A_.q;
        }
        return *(const uint4*)(xq + (g - Dd));
      };
      a.u[0] = grp(c0);
      a.u[1] = grp(c0 + 16);
      v16h bf[4];
#pragma unroll
      for (int t = 0; t < 4; ++t)
        bf[t] = load_wp_frag(Wp, KB, nb_base + t, kb, lane);
#pragma unroll
      for (int t = 0; t < 4; ++t) acc[t] = wmma16(a.v, bf[t], acc[t]);
    }
#pragma unroll
    for (int t = 0; t < 4; ++t) {
      int col = (nb_base + t) * 16 + (lane & 15);
      float bv = bias[col];
#pragma unroll
      for (int r = 0; r < 8; ++r) {
        float v = acc[t][r] + bv;
        v = (v > 0.f) ? v : 0.2f * v;  // leaky relu
        best[t][r] = fmaxf(best[t][r], v);
      }
    }
  }
  int half_off = (lane < 16) ? 0 : 8;
#pragma unroll
  for (int t = 0; t < 4; ++t) {
    int col = (nb_base + t) * 16 + (lane & 15);
#pragma unroll
    for (int r = 0; r < 8; ++r) {
      int rr = row_base + r + half_off;
      Out[(size_t)rr * ldc + col_off + col] = (_Float16)best[t][r];
    }
  }
}

// ---- elementwise helpers ----------------------------------------------------

__global__ void k_transpose_in(const float* __restrict__ in,
                               float* __restrict__ out) {
  size_t i = blockIdx.x * (size_t)blockDim.x + threadIdx.x;
  size_t total = (size_t)Bb * Dd * Nn;
  if (i >= total) return;
  int n = (int)(i % Nn);
  int d = (int)((i / Nn) % Dd);
  int b = (int)(i / ((size_t)Nn * Dd));
  out[((size_t)b * Nn + n) * Dd + d] = in[i];
}

__global__ __launch_bounds__(256) void k_ln(const float* __restrict__ X,
                                            const float* __restrict__ g,
                                            const float* __restrict__ bi,
                                            _Float16* __restrict__ Y,
                                            int rows, int C) {
  int wave = threadIdx.x >> 5, lane = threadIdx.x & 31;
  int row = blockIdx.x * 8 + wave;
  if (row >= rows) return;
  const float* x = X + (size_t)row * C;
  float s = 0.f, s2 = 0.f;
  for (int c = lane; c < C; c += 32) { float v = x[c]; s += v; s2 += v * v; }
  for (int mm = 1; mm <= 16; mm <<= 1) {
    s += __shfl_xor(s, mm, 32);
    s2 += __shfl_xor(s2, mm, 32);
  }
  float mean = s / C;
  float var = s2 / C - mean * mean;
  float rstd = rsqrtf(var + 1e-5f);
  _Float16* y = Y + (size_t)row * C;
  for (int c = lane; c < C; c += 32)
    y[c] = (_Float16)((x[c] - mean) * rstd * g[c] + bi[c]);
}

__global__ void k_build_fuse(const float* __restrict__ pos,
                             const float* __restrict__ guide,
                             _Float16* __restrict__ out) {
  size_t i = blockIdx.x * (size_t)blockDim.x + threadIdx.x;
  size_t total = (size_t)M * FIN_KP;
  if (i >= total) return;
  int c = (int)(i % FIN_KP);
  size_t rn = i / FIN_KP;
  int n = (int)(rn % Nn);
  int b = (int)(rn / Nn);
  float v = 0.f;
  if (c < 3) v = pos[((size_t)b * 3 + c) * Nn + n];
  else if (c < FIN) v = guide[(size_t)b * Gg + (c - 3)];
  out[i] = (_Float16)v;
}

__global__ void k_add(float* __restrict__ x, const float* __restrict__ y) {
  size_t i = blockIdx.x * (size_t)blockDim.x + threadIdx.x;
  size_t total = (size_t)M * Dd;
  if (i >= total) return;
  x[i] += y[i];
}

__global__ void k_final(const float* __restrict__ x,
                        const float* __restrict__ mo,
                        float* __restrict__ out) {
  size_t i = blockIdx.x * (size_t)blockDim.x + threadIdx.x;
  size_t total = (size_t)Bb * Dd * Nn;
  if (i >= total) return;
  int n = (int)(i % Nn);
  int d = (int)((i / Nn) % Dd);
  int b = (int)(i / ((size_t)Nn * Dd));
  size_t r = ((size_t)b * Nn + n) * Dd + d;
  out[i] = x[r] + mo[r];
}

// ---- host helpers -----------------------------------------------------------

struct WS {
  char* p;
  template <class T>
  T* take(size_t elems) {
    char* r = p;
    size_t bytes = (elems * sizeof(T) + 255) & ~(size_t)255;
    p += bytes;
    return (T*)r;
  }
};

void launch_pack(hipStream_t s, const float* W, _Float16* Wp, int K, int Nc,
                 int Kp, int Np) {
  int total = (Np / 16) * (Kp / 32) * 512;
  k_pack_w<<<(total + 255) / 256, 256, 0, s>>>(W, Wp, K, Nc, Kp, Np);
}

void launch_gemm(hipStream_t s, const _Float16* A, int lda, const _Float16* Wp,
                 int Kp, const float* bias, float* Cf, _Float16* Ch, int ldc,
                 int col_off, int Mrows, int Np, int Nreal, int act) {
  dim3 g(Np / 64, Mrows / 128);
  k_gemm<<<g, 256, 0, s>>>(A, lda, Wp, Kp >> 5, bias, Cf, Ch, ldc, col_off,
                           Mrows, Nreal, act);
}

}  // namespace

extern "C" void kernel_launch(void* const* d_in, const int* in_sizes, int n_in,
                              void* d_out, int out_size, void* d_ws,
                              size_t ws_size, hipStream_t stream) {
  (void)in_sizes; (void)n_in; (void)out_size; (void)ws_size;
  const float* q_in      = (const float*)d_in[0];
  const float* in_pos    = (const float*)d_in[1];
  const float* guide     = (const float*)d_in[2];
  const float* norm1_g   = (const float*)d_in[3];
  const float* norm1_b   = (const float*)d_in[4];
  const float* qkv_w     = (const float*)d_in[5];
  const float* sa_proj_w = (const float*)d_in[6];
  const float* sa_proj_b = (const float*)d_in[7];
  const float* normq_g   = (const float*)d_in[8];
  const float* normq_b   = (const float*)d_in[9];
  const float* fuse1_w   = (const float*)d_in[10];
  const float* fuse1_b   = (const float*)d_in[11];
  const float* fuse2_w   = (const float*)d_in[12];
  const float* fuse2_b   = (const float*)d_in[13];
  const float* normv_g   = (const float*)d_in[14];
  const float* normv_b   = (const float*)d_in[15];
  const float* ca_q_w    = (const float*)d_in[16];
  const float* ca_k_w    = (const float*)d_in[17];
  const float* ca_v_w    = (const float*)d_in[18];
  const float* ca_proj_w = (const float*)d_in[19];
  const float* ca_proj_b = (const float*)d_in[20];
  const float* norm2_g   = (const float*)d_in[21];
  const float* norm2_b   = (const float*)d_in[22];
  const float* mlp1_w    = (const float*)d_in[23];
  const float* mlp1_b    = (const float*)d_in[24];
  const float* mlp2_w    = (const float*)d_in[25];
  const float* mlp2_b    = (const float*)d_in[26];
  const float* knn_w     = (const float*)d_in[27];
  const float* knn_b     = (const float*)d_in[28];
  const float* merge_w   = (const float*)d_in[29];
  const float* merge_b   = (const float*)d_in[30];
  const float* knnx_w    = (const float*)d_in[31];
  const float* knnx_b    = (const float*)d_in[32];
  const float* mergex_w  = (const float*)d_in[33];
  const float* mergex_b  = (const float*)d_in[34];
  const int* self_idx    = (const int*)d_in[35];
  const int* cross_idx   = (const int*)d_in[36];
  float* out = (float*)d_out;

  WS ws{(char*)d_ws};
  // packed weights (f16)
  _Float16* wp_qkv    = ws.take<_Float16>((size_t)Dd * 3 * Dd);
  _Float16* wp_sa     = ws.take<_Float16>((size_t)Dd * Dd);
  _Float16* wp_fuse1  = ws.take<_Float16>((size_t)FIN_KP * FIN_NP);
  _Float16* wp_fuse2  = ws.take<_Float16>((size_t)FIN_KP * Dd);
  _Float16* wp_caq    = ws.take<_Float16>((size_t)Dd * Dd);
  _Float16* wp_cak    = ws.take<_Float16>((size_t)Dd * Dd);
  _Float16* wp_cav    = ws.take<_Float16>((size_t)Dd * Dd);
  _Float16* wp_caproj = ws.take<_Float16>((size_t)Dd * Dd);
  _Float16* wp_mlp1   = ws.take<_Float16>((size_t)Dd * HID);
  _Float16* wp_mlp2   = ws.take<_Float16>((size_t)HID * Dd);
  _Float16* wp_knn    = ws.take<_Float16>((size_t)2 * Dd * Dd);
  _Float16* wp_merge  = ws.take<_Float16>((size_t)2 * Dd * Dd);
  _Float16* wp_knnx   = ws.take<_Float16>((size_t)2 * Dd * Dd);
  _Float16* wp_mergex = ws.take<_Float16>((size_t)2 * Dd * Dd);
  // activations
  float*    xf     = ws.take<float>((size_t)M * Dd);       // running q [B*N, D]
  _Float16* nq16   = ws.take<_Float16>((size_t)M * Dd);    // LN outs (reused)
  _Float16* qkv16  = ws.take<_Float16>((size_t)M * 3 * Dd);// qkv; reused qh/kh/vh
  _Float16* attn16 = ws.take<_Float16>((size_t)M * Dd);    // attn out
  _Float16* mrg16  = ws.take<_Float16>((size_t)M * 2 * Dd);// concat buffer
  float*    tmpf   = ws.take<float>((size_t)M * Dd);       // q_1 / q_2 / mlp out
  _Float16* fin16  = ws.take<_Float16>((size_t)M * FIN_KP);
  _Float16* f1o16  = ws.take<_Float16>((size_t)M * FIN_NP);
  float*    vbuf   = ws.take<float>((size_t)M * Dd);
  _Float16* nq2_16 = ws.take<_Float16>((size_t)M * Dd);
  _Float16* nv16   = ws.take<_Float16>((size_t)M * Dd);
  _Float16* m16    = ws.take<_Float16>((size_t)M * HID);

  const float scale = 0.14433756729740643f;  // 48^-0.5

  // 1. pack all weights to WMMA fragment order
  launch_pack(stream, qkv_w,     wp_qkv,    Dd, 3 * Dd, Dd, 3 * Dd);
  launch_pack(stream, sa_proj_w, wp_sa,     Dd, Dd, Dd, Dd);
  launch_pack(stream, fuse1_w,   wp_fuse1,  FIN, FIN, FIN_KP, FIN_NP);
  launch_pack(stream, fuse2_w,   wp_fuse2,  FIN, Dd, FIN_KP, Dd);
  launch_pack(stream, ca_q_w,    wp_caq,    Dd, Dd, Dd, Dd);
  launch_pack(stream, ca_k_w,    wp_cak,    Dd, Dd, Dd, Dd);
  launch_pack(stream, ca_v_w,    wp_cav,    Dd, Dd, Dd, Dd);
  launch_pack(stream, ca_proj_w, wp_caproj, Dd, Dd, Dd, Dd);
  launch_pack(stream, mlp1_w,    wp_mlp1,   Dd, HID, Dd, HID);
  launch_pack(stream, mlp2_w,    wp_mlp2,   HID, Dd, HID, Dd);
  launch_pack(stream, knn_w,     wp_knn,    2 * Dd, Dd, 2 * Dd, Dd);
  launch_pack(stream, merge_w,   wp_merge,  2 * Dd, Dd, 2 * Dd, Dd);
  launch_pack(stream, knnx_w,    wp_knnx,   2 * Dd, Dd, 2 * Dd, Dd);
  launch_pack(stream, mergex_w,  wp_mergex, 2 * Dd, Dd, 2 * Dd, Dd);

  // 2. q [B,D,N] -> x [B*N,D]
  {
    size_t tot = (size_t)Bb * Dd * Nn;
    k_transpose_in<<<(tot + 255) / 256, 256, 0, stream>>>(q_in, xf);
  }
  // 3. norm1 -> f16
  k_ln<<<M / 8, 256, 0, stream>>>(xf, norm1_g, norm1_b, nq16, M, Dd);
  // 4. qkv GEMM (no bias)
  launch_gemm(stream, nq16, Dd, wp_qkv, Dd, nullptr, nullptr, qkv16, 3 * Dd, 0,
              M, 3 * Dd, 3 * Dd, 0);
  // 5. self attention
  k_attn<<<dim3(Bb * Hh, Nn / 128), 256, 0, stream>>>(
      qkv16, 3 * Dd, 0, qkv16, 3 * Dd, Dd, qkv16, 3 * Dd, 2 * Dd, attn16, Dd,
      Nn, Nn, scale);
  // 6. sa proj -> concat buf cols [0,384)
  launch_gemm(stream, attn16, Dd, wp_sa, Dd, sa_proj_b, nullptr, mrg16, 2 * Dd,
              0, M, Dd, Dd, 0);
  // 7. self knn branch -> concat buf cols [384,768)
  k_knn<<<dim3(Dd / 64, M / 128), 256, 0, stream>>>(nq16, nq16, self_idx,
                                                    wp_knn, knn_b, mrg16,
                                                    2 * Dd, Dd);
  // 8. merge GEMM -> q_1 (f32), 9. residual
  launch_gemm(stream, mrg16, 2 * Dd, wp_merge, 2 * Dd, merge_b, tmpf, nullptr,
              Dd, 0, M, Dd, Dd, 0);
  {
    size_t tot = (size_t)M * Dd;
    k_add<<<(tot + 255) / 256, 256, 0, stream>>>(xf, tmpf);
  }
  // 10. fuse branch: v = gelu([pos,guide] @ f1) @ f2
  {
    size_t tot = (size_t)M * FIN_KP;
    k_build_fuse<<<(tot + 255) / 256, 256, 0, stream>>>(in_pos, guide, fin16);
  }
  launch_gemm(stream, fin16, FIN_KP, wp_fuse1, FIN_KP, fuse1_b, nullptr, f1o16,
              FIN_NP, 0, M, FIN_NP, FIN, 1);
  launch_gemm(stream, f1o16, FIN_NP, wp_fuse2, FIN_KP, fuse2_b, vbuf, nullptr,
              Dd, 0, M, Dd, Dd, 0);
  // 11-12. norms
  k_ln<<<M / 8, 256, 0, stream>>>(xf, normq_g, normq_b, nq2_16, M, Dd);
  k_ln<<<M / 8, 256, 0, stream>>>(vbuf, normv_g, normv_b, nv16, M, Dd);
  // 13. cross q/k/v projections (reuse qkv16 region)
  _Float16* qh16 = qkv16;
  _Float16* kh16 = qkv16 + (size_t)M * Dd;
  _Float16* vh16 = qkv16 + (size_t)2 * M * Dd;
  launch_gemm(stream, nq2_16, Dd, wp_caq, Dd, nullptr, nullptr, qh16, Dd, 0, M,
              Dd, Dd, 0);
  launch_gemm(stream, nv16, Dd, wp_cak, Dd, nullptr, nullptr, kh16, Dd, 0, M,
              Dd, Dd, 0);
  launch_gemm(stream, nv16, Dd, wp_cav, Dd, nullptr, nullptr, vh16, Dd, 0, M,
              Dd, Dd, 0);
  // 14. cross attention
  k_attn<<<dim3(Bb * Hh, Nn / 128), 256, 0, stream>>>(
      qh16, Dd, 0, kh16, Dd, 0, vh16, Dd, 0, attn16, Dd, Nn, Nn, scale);
  // 15. ca proj -> concat cols [0,384)
  launch_gemm(stream, attn16, Dd, wp_caproj, Dd, ca_proj_b, nullptr, mrg16,
              2 * Dd, 0, M, Dd, Dd, 0);
  // 16. cross knn (gather from norm_v, query rows norm_q)
  k_knn<<<dim3(Dd / 64, M / 128), 256, 0, stream>>>(nv16, nq2_16, cross_idx,
                                                    wp_knnx, knnx_b, mrg16,
                                                    2 * Dd, Dd);
  // 17. mergex GEMM -> q_2 + residual
  launch_gemm(stream, mrg16, 2 * Dd, wp_mergex, 2 * Dd, mergex_b, tmpf, nullptr,
              Dd, 0, M, Dd, Dd, 0);
  {
    size_t tot = (size_t)M * Dd;
    k_add<<<(tot + 255) / 256, 256, 0, stream>>>(xf, tmpf);
  }
  // 18-20. MLP: h = LN(x); x += gelu(h@W1+b1)@W2+b2
  k_ln<<<M / 8, 256, 0, stream>>>(xf, norm2_g, norm2_b, nq16, M, Dd);
  launch_gemm(stream, nq16, Dd, wp_mlp1, Dd, mlp1_b, nullptr, m16, HID, 0, M,
              HID, HID, 1);
  launch_gemm(stream, m16, HID, wp_mlp2, HID, mlp2_b, tmpf, nullptr, Dd, 0, M,
              Dd, Dd, 0);
  // 21. final residual + transpose out to [B,D,N]
  {
    size_t tot = (size_t)Bb * Dd * Nn;
    k_final<<<(tot + 255) / 256, 256, 0, stream>>>(xf, tmpf, out);
  }
}
